// MultiHeadAttention_27127013441542
// MI455X (gfx1250) — matrix operand order repaired and checked by reading, compiled, not run on hardware
//
#include <hip/hip_runtime.h>
#include <hip/hip_bf16.h>
#include <stdint.h>

typedef __attribute__((ext_vector_type(16))) _Float16     v16h;
typedef __attribute__((ext_vector_type(8)))  float        v8f;
typedef __attribute__((ext_vector_type(4)))  unsigned int u32x4;
typedef __attribute__((ext_vector_type(8)))  int          i32x8;
typedef __attribute__((ext_vector_type(4)))  int          i32x4;

#define B_  4
#define N_  2048
#define M_  2048
#define D_  1024
#define H_  16
#define HD_ 64

#if defined(__HIP_DEVICE_COMPILE__) && defined(__has_builtin)
#  if __has_builtin(__builtin_amdgcn_tensor_load_to_lds)
#    define HAVE_TDM 1
#  else
#    define HAVE_TDM 0
#  endif
#else
#  define HAVE_TDM 0
#endif

#if HAVE_TDM
#  if __has_builtin(__builtin_amdgcn_s_wait_tensorcnt)
#    define WAIT_TENSORCNT(n) __builtin_amdgcn_s_wait_tensorcnt((short)(n))
#  else
#    define WAIT_TENSORCNT(n) asm volatile("s_wait_tensorcnt %0" :: "i"(n) : "memory")
#  endif
#endif

// ---- WMMA fragment index maps (CDNA5 ISA 7.12.2, 16-bit operands, wave32) ----
// A-matrix 16xK: lane -> M = lane%16, half = lane/16; element e -> K index:
__device__ __forceinline__ int kmapA(int e, int hf) {
    return (hf << 4) + e;
}
// B-matrix Kx16: lane -> N = lane%16; lanes 0-15 hold K=0..15, lanes 16-31 K=16..31
__device__ __forceinline__ int kmapB(int e, int hf) {
    return (hf << 4) + e;
}

__device__ __forceinline__ v8f wmma16(v16h a, v16h b, v8f c) {
    return __builtin_amdgcn_wmma_f32_16x16x32_f16(
        /*neg_a=*/false, a, /*neg_b=*/false, b,
        /*c_mod=*/(short)0, c, /*reuse_a=*/false, /*reuse_b=*/false);
}

// A operand from row-major [rows][ld], rows indexed by lane%16
__device__ __forceinline__ v16h fragA_f32(const float* base, long ld, int k0, int lane) {
    int m = lane & 15, hf = lane >> 4;
    v16h a;
#pragma unroll
    for (int e = 0; e < 16; ++e)
        a[e] = (_Float16)base[(long)m * ld + k0 + kmapA(e, hf)];
    return a;
}
__device__ __forceinline__ v16h fragA_f16(const _Float16* base, long ld, int k0, int lane) {
    int m = lane & 15, hf = lane >> 4;
    v16h a;
#pragma unroll
    for (int e = 0; e < 16; ++e)
        a[e] = base[(long)m * ld + k0 + kmapA(e, hf)];
    return a;
}
// B operand where storage is [N][K] row-major (per output column n, K contiguous)
__device__ __forceinline__ v16h fragB_nk_f16(const _Float16* base, long ld, int k0, int lane) {
    int n = lane & 15, hf = lane >> 4;
    v16h b;
#pragma unroll
    for (int e = 0; e < 16; ++e)
        b[e] = base[(long)n * ld + k0 + kmapB(e, hf)];
    return b;
}
// B operands from an LDS tile of shape [32 rows][HD_ halves]
__device__ __forceinline__ v16h fragB_nk_lds(const _Float16* tile, int row0, int k0, int lane) {
    int n = lane & 15, hf = lane >> 4;
    v16h b;
#pragma unroll
    for (int e = 0; e < 16; ++e)
        b[e] = tile[(row0 + n) * HD_ + k0 + kmapB(e, hf)];   // 32B contiguous per lane
    return b;
}
__device__ __forceinline__ v16h fragB_kn_lds(const _Float16* tile, int n0, int lane) {
    int n = lane & 15, hf = lane >> 4;
    v16h b;
#pragma unroll
    for (int e = 0; e < 16; ++e)
        b[e] = tile[kmapB(e, hf) * HD_ + n0 + n];
    return b;
}

#if HAVE_TDM
// Issue one TDM 2-D tile load: 32 rows x 64 halves from a [rows][D_] f16 tensor.
__device__ __forceinline__ void tdm_load_tile(unsigned lds_off, unsigned long long gaddr, i32x8 g1) {
    u32x4 g0;
    g0.x = 1u;                                                    // count=1, user mode
    g0.y = lds_off;                                               // lds_addr (bytes)
    g0.z = (unsigned)(gaddr & 0xFFFFFFFFull);                     // global_addr[31:0]
    g0.w = (unsigned)((gaddr >> 32) & 0x01FFFFFFull) | (2u << 30);// global_addr[56:32] | type=2
    i32x4 z4 = {0, 0, 0, 0};
#if __clang_major__ >= 23
    i32x8 z8 = {0, 0, 0, 0, 0, 0, 0, 0};
    __builtin_amdgcn_tensor_load_to_lds(g0, g1, z4, z4, z8, 0);
#else
    __builtin_amdgcn_tensor_load_to_lds(g0, g1, z4, z4, 0);
#endif
}
#endif

// ---------------- weight conversion: f32 -> f16, vectorized x4 ----------------
__global__ void __launch_bounds__(256)
cvt_f32_f16(const float* __restrict__ s, _Float16* __restrict__ d, int n4)
{
    int i = blockIdx.x * 256 + threadIdx.x;
    if (i < n4) {
        float4 v = ((const float4*)s)[i];
        d[4 * i + 0] = (_Float16)v.x;
        d[4 * i + 1] = (_Float16)v.y;
        d[4 * i + 2] = (_Float16)v.z;
        d[4 * i + 3] = (_Float16)v.w;
    }
}

// ---------------- Projection GEMM: Y(f16) = X(f32) @ Wh^T + bias ----------------
// one wave = one 32x64 C tile (2x4 WMMA accumulators), K loop of 32
__global__ void __launch_bounds__(256)
proj_gemm(const float* __restrict__ X, const _Float16* __restrict__ Wh,
          const float* __restrict__ bias, _Float16* __restrict__ Y, int rows)
{
    const int lane = threadIdx.x & 31;
    const int wid  = blockIdx.x * 8 + (threadIdx.x >> 5);
    const int TJ   = D_ / 64;
    const int j0   = (wid % TJ) * 64;
    const int i0   = (wid / TJ) * 32;
    if (i0 >= rows) return;

    v8f acc[2][4] = {};
    const float* A0 = X + (long)i0 * D_;
    const float* A1 = X + (long)(i0 + 16) * D_;
    for (int k0 = 0; k0 < D_; k0 += 32) {
        v16h a0 = fragA_f32(A0, D_, k0, lane);
        v16h a1 = fragA_f32(A1, D_, k0, lane);
#pragma unroll
        for (int j = 0; j < 4; ++j) {
            v16h bb = fragB_nk_f16(Wh + (long)(j0 + 16 * j) * D_, D_, k0, lane);
            acc[0][j] = wmma16(a0, bb, acc[0][j]);
            acc[1][j] = wmma16(a1, bb, acc[1][j]);
        }
    }
    const int n = lane & 15, hf = lane >> 4;
#pragma unroll
    for (int j = 0; j < 4; ++j) {
        float bj = bias[j0 + 16 * j + n];
#pragma unroll
        for (int r = 0; r < 8; ++r) {
            int row = r + 8 * hf;
            Y[(long)(i0 + row) * D_ + j0 + 16 * j + n]      = (_Float16)(acc[0][j][r] + bj);
            Y[(long)(i0 + 16 + row) * D_ + j0 + 16 * j + n] = (_Float16)(acc[1][j][r] + bj);
        }
    }
}

// ---------------- Output GEMM: out(f32) = O(f16) @ Woh^T + bo ----------------
__global__ void __launch_bounds__(256)
out_gemm(const _Float16* __restrict__ X, const _Float16* __restrict__ Wh,
         const float* __restrict__ bias, float* __restrict__ Y, int rows)
{
    const int lane = threadIdx.x & 31;
    const int wid  = blockIdx.x * 8 + (threadIdx.x >> 5);
    const int TJ   = D_ / 64;
    const int j0   = (wid % TJ) * 64;
    const int i0   = (wid / TJ) * 32;
    if (i0 >= rows) return;

    v8f acc[2][4] = {};
    const _Float16* A0 = X + (long)i0 * D_;
    const _Float16* A1 = X + (long)(i0 + 16) * D_;
    for (int k0 = 0; k0 < D_; k0 += 32) {
        v16h a0 = fragA_f16(A0, D_, k0, lane);
        v16h a1 = fragA_f16(A1, D_, k0, lane);
#pragma unroll
        for (int j = 0; j < 4; ++j) {
            v16h bb = fragB_nk_f16(Wh + (long)(j0 + 16 * j) * D_, D_, k0, lane);
            acc[0][j] = wmma16(a0, bb, acc[0][j]);
            acc[1][j] = wmma16(a1, bb, acc[1][j]);
        }
    }
    const int n = lane & 15, hf = lane >> 4;
#pragma unroll
    for (int j = 0; j < 4; ++j) {
        float bj = bias[j0 + 16 * j + n];
#pragma unroll
        for (int r = 0; r < 8; ++r) {
            int row = r + 8 * hf;
            Y[(long)(i0 + row) * D_ + j0 + 16 * j + n]      = acc[0][j][r] + bj;
            Y[(long)(i0 + 16 + row) * D_ + j0 + 16 * j + n] = acc[1][j][r] + bj;
        }
    }
}

// ---------------- Flash attention: one wave per (b, h, 16 query rows) ----------------
// K/V tiles (32 keys x 64 dims, f16) staged into LDS by the Tensor Data Mover,
// double-buffered on TENSORcnt so DMA overlaps WMMA+softmax of the previous tile.
#define AW 2

__global__ void __launch_bounds__(AW * 32)
attn_kernel(const _Float16* __restrict__ qh, const _Float16* __restrict__ kh,
            const _Float16* __restrict__ vh, const float* __restrict__ bias,
            const unsigned char* __restrict__ mask, _Float16* __restrict__ Oh)
{
    __shared__ _Float16 sK[AW][2][32 * HD_];
    __shared__ _Float16 sV[AW][2][32 * HD_];
    __shared__ _Float16 sP[AW][16 * 32];

    const int lane = threadIdx.x & 31;
    const int w    = __builtin_amdgcn_readfirstlane((int)(threadIdx.x >> 5));
    const int wid  = blockIdx.x * AW + w;
    const int QT = N_ / 16;
    const int qt = wid % QT;
    const int h  = (wid / QT) % H_;
    const int b  = wid / (QT * H_);
    const int q0 = qt * 16;
    const int n = lane & 15, hf = lane >> 4;

    // Q fragments: loaded once, reused for all 64 key iterations (HD=64 -> two K=32 frags)
    const _Float16* qbase = qh + ((long)b * N_ + q0) * D_ + h * HD_;
    v16h qa0 = fragA_f16(qbase, D_, 0,  lane);
    v16h qa1 = fragA_f16(qbase, D_, 32, lane);

    float mi[8], li[8];
    v8f o0 = {}, o1 = {}, o2 = {}, o3 = {};
#pragma unroll
    for (int r = 0; r < 8; ++r) { mi[r] = -1e30f; li[r] = 0.f; }

    const float scale = 0.125f;   // 1/sqrt(HD=64), exact
    const float* biasb = bias + (long)b * N_ * M_;
    const unsigned char* maskb = mask + (long)b * M_;

    const _Float16* kg = kh + ((long)b * M_) * D_ + h * HD_;
    const _Float16* vg = vh + ((long)b * M_) * D_ + h * HD_;

#if HAVE_TDM
    // D# group1 (shared by K and V tiles): data_size=2B; tensor dim0=D_, dim1=B_*M_;
    // tile 64 x 32; dim0 stride = D_ elements.
    i32x8 g1 = { (int)(1u << 16),
                 (int)((unsigned)D_ << 16),
                 (int)((unsigned)(B_ * M_) << 16),
                 (int)((unsigned)HD_ << 16),
                 32,
                 D_,
                 0, 0 };
    const unsigned long long kga = (unsigned long long)(const void*)kg;
    const unsigned long long vga = (unsigned long long)(const void*)vg;
    unsigned ldsK[2] = { (unsigned)(uintptr_t)&sK[w][0][0], (unsigned)(uintptr_t)&sK[w][1][0] };
    unsigned ldsV[2] = { (unsigned)(uintptr_t)&sV[w][0][0], (unsigned)(uintptr_t)&sV[w][1][0] };
    const unsigned long long tileBytes = (unsigned long long)32 * D_ * 2;  // 32 rows down the tensor
    tdm_load_tile(ldsK[0], kga, g1);
    tdm_load_tile(ldsV[0], vga, g1);
    tdm_load_tile(ldsK[1], kga + tileBytes, g1);
    tdm_load_tile(ldsV[1], vga + tileBytes, g1);
#endif

    const int NIT = M_ / 32;
    for (int it = 0; it < NIT; ++it) {
        const int buf = it & 1;
        const int kb  = it * 32;
#if HAVE_TDM
        WAIT_TENSORCNT(2);       // oldest (this iteration's) K+V pair has landed
#else
        {   // fallback staging: coalesced u64 copies, 16 per lane per tile
            const unsigned long long* srcK = (const unsigned long long*)(kg + (long)kb * D_);
            const unsigned long long* srcV = (const unsigned long long*)(vg + (long)kb * D_);
            unsigned long long* dK = (unsigned long long*)&sK[w][buf][0];
            unsigned long long* dV = (unsigned long long*)&sV[w][buf][0];
            for (int t = lane; t < 512; t += 32) {
                int row = t >> 4, col = t & 15;
                dK[t] = srcK[(long)row * (D_ / 4) + col];
                dV[t] = srcV[(long)row * (D_ / 4) + col];
            }
            asm volatile("s_wait_dscnt 0x0" ::: "memory");
        }
#endif
        const _Float16* Kt = &sK[w][buf][0];
        const _Float16* Vt = &sV[w][buf][0];

        // ---- S = (q*scale) @ k^T for 32 keys: two 16x16 C tiles ----
        v8f s0 = {}, s1 = {};
        {
            v16h b00 = fragB_nk_lds(Kt, 0,  0,  lane);
            v16h b01 = fragB_nk_lds(Kt, 0,  32, lane);
            s0 = wmma16(qa0, b00, s0);
            s0 = wmma16(qa1, b01, s0);
            v16h b10 = fragB_nk_lds(Kt, 16, 0,  lane);
            v16h b11 = fragB_nk_lds(Kt, 16, 32, lane);
            s1 = wmma16(qa0, b10, s1);
            s1 = wmma16(qa1, b11, s1);
        }
        // ---- scale + bias + key mask ----
        const float pen0 = maskb[kb + n]      ? -1e30f : 0.f;
        const float pen1 = maskb[kb + 16 + n] ? -1e30f : 0.f;
#pragma unroll
        for (int r = 0; r < 8; ++r) {
            long row = q0 + r + 8 * hf;
            s0[r] = s0[r] * scale + biasb[row * M_ + kb + n]      + pen0;
            s1[r] = s1[r] * scale + biasb[row * M_ + kb + 16 + n] + pen1;
        }
        // ---- online softmax (row r+8*hf: its 16 columns live across a 16-lane group) ----
#pragma unroll
        for (int r = 0; r < 8; ++r) {
            float mx = fmaxf(s0[r], s1[r]);
#pragma unroll
            for (int off = 8; off > 0; off >>= 1)
                mx = fmaxf(mx, __shfl_xor(mx, off, 16));
            float mnew  = fmaxf(mi[r], mx);
            float alpha = __expf(mi[r] - mnew);
            mi[r] = mnew;
            float p0 = __expf(s0[r] - mnew);
            float p1 = __expf(s1[r] - mnew);
            float rs = p0 + p1;
#pragma unroll
            for (int off = 8; off > 0; off >>= 1)
                rs += __shfl_xor(rs, off, 16);
            li[r] = li[r] * alpha + rs;
            o0[r] *= alpha; o1[r] *= alpha; o2[r] *= alpha; o3[r] *= alpha;
            int row = r + 8 * hf;
            sP[w][row * 32 + n]      = (_Float16)p0;
            sP[w][row * 32 + 16 + n] = (_Float16)p1;
        }
        // cross-lane transpose through LDS: wait for DS stores before re-reading
        asm volatile("s_wait_dscnt 0x0" ::: "memory");
        v16h pa;
#pragma unroll
        for (int e = 0; e < 16; ++e)
            pa[e] = sP[w][n * 32 + kmapA(e, hf)];   // A-row m = lane%16 (== n)
        // ---- O += P @ V (K=32 keys, 4 d-tiles of 16) ----
        o0 = wmma16(pa, fragB_kn_lds(Vt, 0,  lane), o0);
        o1 = wmma16(pa, fragB_kn_lds(Vt, 16, lane), o1);
        o2 = wmma16(pa, fragB_kn_lds(Vt, 32, lane), o2);
        o3 = wmma16(pa, fragB_kn_lds(Vt, 48, lane), o3);
#if HAVE_TDM
        if (it + 2 < NIT) {
            // this buffer's LDS reads are complete; refill it two tiles ahead
            asm volatile("s_wait_dscnt 0x0" ::: "memory");
            unsigned long long off = (unsigned long long)(kb + 64) * D_ * 2;
            tdm_load_tile(ldsK[buf], kga + off, g1);
            tdm_load_tile(ldsV[buf], vga + off, g1);
        }
#endif
    }
    // ---- normalize and store O as f16 [B*N, D] ----
#pragma unroll
    for (int r = 0; r < 8; ++r) {
        float inv = 1.0f / li[r];
        long row = (long)b * N_ + q0 + r + 8 * hf;
        _Float16* dst = Oh + row * D_ + h * HD_;
        dst[n]      = (_Float16)(o0[r] * inv);
        dst[16 + n] = (_Float16)(o1[r] * inv);
        dst[32 + n] = (_Float16)(o2[r] * inv);
        dst[48 + n] = (_Float16)(o3[r] * inv);
    }
}

extern "C" void kernel_launch(void* const* d_in, const int* in_sizes, int n_in,
                              void* d_out, int out_size, void* d_ws, size_t ws_size,
                              hipStream_t stream)
{
    (void)in_sizes; (void)n_in; (void)out_size; (void)ws_size;
    const float* Q  = (const float*)d_in[0];
    const float* K  = (const float*)d_in[1];
    const float* V  = (const float*)d_in[2];
    const float* attn_bias   = (const float*)d_in[3];
    const unsigned char* msk = (const unsigned char*)d_in[4];
    const float* Wq = (const float*)d_in[5];
    const float* bq = (const float*)d_in[6];
    const float* Wk = (const float*)d_in[7];
    const float* bk = (const float*)d_in[8];
    const float* Wv = (const float*)d_in[9];
    const float* bv = (const float*)d_in[10];
    const float* Wo = (const float*)d_in[11];
    const float* bo = (const float*)d_in[12];
    float* out = (float*)d_out;

    const long tok = (long)B_ * N_;                 // 8192 rows
    _Float16* qh  = (_Float16*)d_ws;                // 16 MB each
    _Float16* kh  = qh + tok * D_;
    _Float16* vh  = kh + tok * D_;
    _Float16* Oh  = vh + tok * D_;
    _Float16* wqh = Oh + tok * D_;                  // 2 MB each
    _Float16* wkh = wqh + (long)D_ * D_;
    _Float16* wvh = wkh + (long)D_ * D_;
    _Float16* woh = wvh + (long)D_ * D_;

    const dim3 blk256(256);
    const int cvt_blocks  = (D_ * D_ / 4) / 256;                   // 1024
    const int gemm_blocks = (int)((tok / 32) * (D_ / 64) / 8);     // 512
    const int attn_blocks = (B_ * H_ * (N_ / 16)) / AW;            // 4096

    hipLaunchKernelGGL(cvt_f32_f16, dim3(cvt_blocks), blk256, 0, stream, Wq, wqh, D_ * D_ / 4);
    hipLaunchKernelGGL(cvt_f32_f16, dim3(cvt_blocks), blk256, 0, stream, Wk, wkh, D_ * D_ / 4);
    hipLaunchKernelGGL(cvt_f32_f16, dim3(cvt_blocks), blk256, 0, stream, Wv, wvh, D_ * D_ / 4);
    hipLaunchKernelGGL(cvt_f32_f16, dim3(cvt_blocks), blk256, 0, stream, Wo, woh, D_ * D_ / 4);

    hipLaunchKernelGGL(proj_gemm, dim3(gemm_blocks), blk256, 0, stream, Q, wqh, bq, qh, (int)tok);
    hipLaunchKernelGGL(proj_gemm, dim3(gemm_blocks), blk256, 0, stream, K, wkh, bk, kh, (int)tok);
    hipLaunchKernelGGL(proj_gemm, dim3(gemm_blocks), blk256, 0, stream, V, wvh, bv, vh, (int)tok);

    hipLaunchKernelGGL(attn_kernel, dim3(attn_blocks), dim3(AW * 32), 0, stream,
                       qh, kh, vh, attn_bias, msk, Oh);

    hipLaunchKernelGGL(out_gemm, dim3(gemm_blocks), blk256, 0, stream, Oh, woh, bo, out, (int)tok);
}
